// RelativeMultiHeadAttn_49984829391397
// MI455X (gfx1250) — matrix operand order repaired and checked by reading, not compiled, not run on hardware
//
#include <hip/hip_runtime.h>
#include <hip/hip_bf16.h>

// ---------------------------------------------------------------------------
// Types / helpers for CDNA5 WMMA (wave32, 16x16x32 bf16 -> f32)
// ---------------------------------------------------------------------------
typedef __attribute__((ext_vector_type(16))) __bf16 v16bf;
typedef __attribute__((ext_vector_type(8)))  float  v8f;

#define D_MODEL 1024
#define N_HEAD  16
#define HEAD_DIM 64
#define BATCH   4
#define MAX_LEN 1024
#define M_TOTAL (BATCH * MAX_LEN)        // 4096
#define R_ROWS  (2 * MAX_LEN + 16)       // padded r rows

__device__ __forceinline__ unsigned short f2bf(float f) {
    union { float f; unsigned u; } x; x.f = f;
    unsigned r = x.u + 0x7FFFu + ((x.u >> 16) & 1u);   // round-to-nearest-even
    return (unsigned short)(r >> 16);
}
__device__ __forceinline__ float bf2f(unsigned short u) {
    union { unsigned u; float f; } x; x.u = ((unsigned)u) << 16; return x.f;
}

__device__ __forceinline__ v8f wmma_bf16(v16bf a, v16bf b, v8f c) {
    return __builtin_amdgcn_wmma_f32_16x16x32_bf16(
        /*neg_a=*/false, a, /*neg_b=*/false, b,
        /*c_mod=*/(short)0, c, /*reuse_a=*/false, /*reuse_b=*/false);
}

// A-fragment (16x32, row-major source, row stride ld elems).
// lane%16 = row M; lane/16 = K-half. elems[0..7]=K(kh*8..+7), elems[8..15]=K(16+kh*8..+7)
__device__ __forceinline__ v16bf load_a_frag(const unsigned short* base, int ld) {
    int lane = threadIdx.x & 31;
    int m = lane & 15, kh = lane >> 4;
    const unsigned short* p = base + (size_t)m * ld + kh * 8;
    union { v16bf v; unsigned short u[16]; } r;
#pragma unroll
    for (int t = 0; t < 8; ++t) { r.u[t] = p[t]; r.u[8 + t] = p[16 + t]; }
    return r.v;
}

// B-fragment where B[k][n] = src[n][k]  (source rows are the N dim, K contiguous)
// lane%16 = col N; lane/16 = K-half; elems[0..7]=src[n][kh*8..+7], elems[8..15]=src[n][16+kh*8..+7]
__device__ __forceinline__ v16bf load_b_colsrc(const unsigned short* base, int ld) {
    int lane = threadIdx.x & 31;
    int n = lane & 15, kh = lane >> 4;
    const unsigned short* p = base + (size_t)n * ld + kh * 8;
    union { v16bf v; unsigned short u[16]; } r;
#pragma unroll
    for (int t = 0; t < 8; ++t) { r.u[t] = p[t]; r.u[8 + t] = p[16 + t]; }
    return r.v;
}

// ---------------------------------------------------------------------------
// Kernel 1a: f32 -> bf16 elementwise convert
// ---------------------------------------------------------------------------
__global__ void cvt_bf16_kernel(const float* __restrict__ in,
                                unsigned short* __restrict__ out, int n) {
    int i = blockIdx.x * blockDim.x + threadIdx.x;
    if (i < n) out[i] = f2bf(in[i]);
}

// ---------------------------------------------------------------------------
// Kernel 1b: f32 [K][N] -> bf16 transposed [N][K]
// ---------------------------------------------------------------------------
__global__ void cvt_bf16_transpose_kernel(const float* __restrict__ in,
                                          unsigned short* __restrict__ out,
                                          int K, int N) {
    int idx = blockIdx.x * blockDim.x + threadIdx.x;
    if (idx >= K * N) return;
    int n = idx / K, k = idx - n * K;            // out index [n][k]
    out[idx] = f2bf(in[(size_t)k * N + n]);
}

// ---------------------------------------------------------------------------
// Kernel 2: r = pos_embed @ Wr  -> bf16 [R_ROWS][64], rows >= 2L zeroed
// ---------------------------------------------------------------------------
__global__ void r_proj_kernel(const float* __restrict__ pos,
                              const float* __restrict__ Wr,
                              unsigned short* __restrict__ rbf) {
    int idx = blockIdx.x * blockDim.x + threadIdx.x;
    if (idx >= R_ROWS * HEAD_DIM) return;
    int l = idx >> 6, d = idx & 63;
    float acc = 0.f;
    if (l < 2 * MAX_LEN) {
#pragma unroll 8
        for (int c = 0; c < HEAD_DIM; ++c)
            acc += pos[l * HEAD_DIM + c] * Wr[c * HEAD_DIM + d];
    }
    rbf[idx] = f2bf(acc);
}

// ---------------------------------------------------------------------------
// Kernel 3: qh = q @ Wq ; write qac = bf16(qh + r_r_bias), qbd = bf16(qh + r_w_bias)
// head-major [B][H][L][hd]. One wave -> 16x64 output tile. Weights pre-transposed.
// ---------------------------------------------------------------------------
__global__ __launch_bounds__(32) void proj_q_kernel(
    const unsigned short* __restrict__ Abf,    // q bf16 [4096][1024]
    const unsigned short* __restrict__ WbfT,   // Wq^T bf16 [1024 N][1024 K]
    const float* __restrict__ r_r_bias, const float* __restrict__ r_w_bias,
    unsigned short* __restrict__ qac, unsigned short* __restrict__ qbd) {
    int mt = blockIdx.x;           // 16-row tile of M
    int nt = blockIdx.y;           // 64-col tile of N
    v8f acc[4] = {};
    for (int k0 = 0; k0 < D_MODEL; k0 += 32) {
        v16bf a = load_a_frag(Abf + (size_t)mt * 16 * D_MODEL + k0, D_MODEL);
#pragma unroll
        for (int j = 0; j < 4; ++j) {
            v16bf b = load_b_colsrc(WbfT + (size_t)(nt * 64 + j * 16) * D_MODEL + k0,
                                    D_MODEL);
            acc[j] = wmma_bf16(a, b, acc[j]);
        }
    }
    int lane = threadIdx.x & 31, col = lane & 15, half = lane >> 4;
#pragma unroll
    for (int j = 0; j < 4; ++j) {
        int N = nt * 64 + j * 16 + col;
        int h = N >> 6, d = N & 63;
        float brr = r_r_bias[h * HEAD_DIM + d];
        float brw = r_w_bias[h * HEAD_DIM + d];
#pragma unroll
        for (int v = 0; v < 8; ++v) {
            int M = mt * 16 + v + half * 8;
            int b_ = M >> 10, l = M & 1023;
            size_t o = (((size_t)b_ * N_HEAD + h) * MAX_LEN + l) * HEAD_DIM + d;
            float val = acc[j][v];
            qac[o] = f2bf(val + brr);
            qbd[o] = f2bf(val + brw);
        }
    }
}

// ---------------------------------------------------------------------------
// Kernel 4: kv = k @ Wkv ; kh bf16 [B][H][L][hd] ; vh stored TRANSPOSED
// [B][H][hd][L] so P@V B-fragments are contiguous. Weights pre-transposed.
// ---------------------------------------------------------------------------
__global__ __launch_bounds__(32) void proj_kv_kernel(
    const unsigned short* __restrict__ Abf,    // k bf16 [4096][1024]
    const unsigned short* __restrict__ WbfT,   // Wkv^T bf16 [2048 N][1024 K]
    unsigned short* __restrict__ khb,          // [B][H][L][hd]
    unsigned short* __restrict__ vhbT) {       // [B][H][hd][L]
    int mt = blockIdx.x;
    int nt = blockIdx.y;           // 0..31 (64-wide tiles over 2048)
    v8f acc[4] = {};
    for (int k0 = 0; k0 < D_MODEL; k0 += 32) {
        v16bf a = load_a_frag(Abf + (size_t)mt * 16 * D_MODEL + k0, D_MODEL);
#pragma unroll
        for (int j = 0; j < 4; ++j) {
            v16bf b = load_b_colsrc(WbfT + (size_t)(nt * 64 + j * 16) * D_MODEL + k0,
                                    D_MODEL);
            acc[j] = wmma_bf16(a, b, acc[j]);
        }
    }
    int lane = threadIdx.x & 31, col = lane & 15, half = lane >> 4;
#pragma unroll
    for (int j = 0; j < 4; ++j) {
        int N = nt * 64 + j * 16 + col;
        int isV = (N >= D_MODEL);
        int Nh = isV ? (N - D_MODEL) : N;
        int h = Nh >> 6, d = Nh & 63;
#pragma unroll
        for (int v = 0; v < 8; ++v) {
            int M = mt * 16 + v + half * 8;
            int b_ = M >> 10, l = M & 1023;
            float val = acc[j][v];
            if (isV) {
                size_t o = (((size_t)b_ * N_HEAD + h) * HEAD_DIM + d) * MAX_LEN + l;
                vhbT[o] = f2bf(val);
            } else {
                size_t o = (((size_t)b_ * N_HEAD + h) * MAX_LEN + l) * HEAD_DIM + d;
                khb[o] = f2bf(val);
            }
        }
    }
}

// ---------------------------------------------------------------------------
// Kernel 5: flash attention with relative-shift band.
// One wave per (b, h, 16-row q tile).  BD_shift[i,j] = BD_raw[i, L + j - i];
// per tile the needed r rows form a band [L-i0-15, 2L-i0] of width L+16.
// ---------------------------------------------------------------------------
#define BANDW (MAX_LEN + 16)   // 1040
// dyn smem: BDu ushort[16*BANDW] | S f32[16*32] | P ushort[16*32] | scale f32[16] | sum f32[16]
#define SM_BD_BYTES   (16 * BANDW * 2)
#define SM_S_OFF      (SM_BD_BYTES)
#define SM_P_OFF      (SM_S_OFF + 16 * 32 * 4)
#define SM_SCALE_OFF  (SM_P_OFF + 16 * 32 * 2)
#define SM_SUM_OFF    (SM_SCALE_OFF + 16 * 4)
#define SM_TOTAL      (SM_SUM_OFF + 16 * 4)

__global__ __launch_bounds__(32) void attn_kernel(
    const unsigned short* __restrict__ qac, const unsigned short* __restrict__ qbd,
    const unsigned short* __restrict__ kh,  const unsigned short* __restrict__ vhT,
    const unsigned short* __restrict__ rbf, const int* __restrict__ mask,
    float* __restrict__ out) {
    extern __shared__ char smem[];
    unsigned short* BDu = (unsigned short*)smem;
    float* S            = (float*)(smem + SM_S_OFF);
    unsigned short* P   = (unsigned short*)(smem + SM_P_OFF);
    float* rowscale     = (float*)(smem + SM_SCALE_OFF);
    float* rowsum       = (float*)(smem + SM_SUM_OFF);

    const int L = MAX_LEN;
    int qt = blockIdx.x, h = blockIdx.y, b = blockIdx.z;
    int i0 = qt * 16;
    int lane = threadIdx.x & 31, col = lane & 15, half = lane >> 4;
    size_t headoff = (((size_t)b * N_HEAD + h) * MAX_LEN) * HEAD_DIM;   // also = vT head base

    // q fragments (held in VGPRs for the whole block)
    v16bf aq0 = load_a_frag(qac + headoff + (size_t)i0 * HEAD_DIM + 0,  HEAD_DIM);
    v16bf aq1 = load_a_frag(qac + headoff + (size_t)i0 * HEAD_DIM + 32, HEAD_DIM);
    v16bf ab0 = load_a_frag(qbd + headoff + (size_t)i0 * HEAD_DIM + 0,  HEAD_DIM);
    v16bf ab1 = load_a_frag(qbd + headoff + (size_t)i0 * HEAD_DIM + 32, HEAD_DIM);

    // -------- BD band: 16 x (L+16) = qbd_tile @ r_band^T --------
    int l0 = L - i0 - 15;   // >= 1
    for (int t = 0; t < BANDW / 16; ++t) {
        const unsigned short* rb = rbf + (size_t)(l0 + t * 16) * HEAD_DIM;
        v16bf b0 = load_b_colsrc(rb + 0,  HEAD_DIM);
        v16bf b1 = load_b_colsrc(rb + 32, HEAD_DIM);
        v8f c = {};
        c = wmma_bf16(ab0, b0, c);
        c = wmma_bf16(ab1, b1, c);
#pragma unroll
        for (int v = 0; v < 8; ++v)
            BDu[(v + half * 8) * BANDW + t * 16 + col] = f2bf(c[v]);
    }
    __syncthreads();

    float m_run = -__builtin_inff();   // valid on lanes 0..15 (row = lane)
    float s_run = 0.f;
    v8f o[4] = {};
    const float inv_scale = 0.125f;    // 1/sqrt(64)

    for (int j0 = 0; j0 < L; j0 += 32) {
        // ---- S = (Q K^T + BD_shift) * inv_scale, masked ----
#pragma unroll
        for (int st = 0; st < 2; ++st) {
            const unsigned short* kb = kh + headoff + (size_t)(j0 + st * 16) * HEAD_DIM;
            v16bf b0 = load_b_colsrc(kb + 0,  HEAD_DIM);
            v16bf b1 = load_b_colsrc(kb + 32, HEAD_DIM);
            v8f c = {};
            c = wmma_bf16(aq0, b0, c);
            c = wmma_bf16(aq1, b1, c);
            int j = j0 + st * 16 + col;
            int mk = mask[b * L + j];
#pragma unroll
            for (int v = 0; v < 8; ++v) {
                int r = v + half * 8;
                float bd = bf2f(BDu[r * BANDW + (j + 15 - r)]);
                float val = (c[v] + bd) * inv_scale;
                S[r * 32 + st * 16 + col] = mk ? val : -__builtin_inff();
            }
        }
        __syncthreads();

        // ---- online softmax (lanes 0..15 own one row each) ----
        if (lane < 16) {
            int r = lane;
            float mx = m_run;
#pragma unroll
            for (int j = 0; j < 32; ++j) mx = fmaxf(mx, S[r * 32 + j]);
            float sc, ssum = 0.f;
            if (mx == -__builtin_inff()) {
                sc = 1.f;
#pragma unroll
                for (int j = 0; j < 32; ++j) P[r * 32 + j] = 0;
            } else {
                sc = __expf(m_run - mx);   // m_run=-inf -> 0
#pragma unroll
                for (int j = 0; j < 32; ++j) {
                    float e = __expf(S[r * 32 + j] - mx);
                    P[r * 32 + j] = f2bf(e);
                    ssum += e;
                }
            }
            s_run = s_run * sc + ssum;
            m_run = mx;
            rowscale[r] = sc;
        }
        __syncthreads();

        // ---- rescale O accumulators ----
        float scl[8];
#pragma unroll
        for (int v = 0; v < 8; ++v) scl[v] = rowscale[v + half * 8];
#pragma unroll
        for (int t = 0; t < 4; ++t)
#pragma unroll
            for (int v = 0; v < 8; ++v) o[t][v] *= scl[v];

        // ---- O += P @ V   (P: 16x32 bf16 A-frag, V^T: [hd][L] -> contiguous B) ----
        v16bf pa = load_a_frag(P, 32);
#pragma unroll
        for (int t = 0; t < 4; ++t) {
            // B[k][n] = V[j0+k][t*16+n] = vT[head][t*16+n][j0+k]
            v16bf vb = load_b_colsrc(vhT + headoff + (size_t)(t * 16) * MAX_LEN + j0,
                                     MAX_LEN);
            o[t] = wmma_bf16(pa, vb, o[t]);
        }
        __syncthreads();   // protect S/P before next iteration
    }

    if (lane < 16) rowsum[lane] = s_run;
    __syncthreads();

#pragma unroll
    for (int t = 0; t < 4; ++t) {
#pragma unroll
        for (int v = 0; v < 8; ++v) {
            int r = v + half * 8;
            float s = rowsum[r];
            float val = (s > 0.f) ? (o[t][v] / s) : 0.f;
            int i = i0 + r;
            out[((size_t)b * MAX_LEN + i) * D_MODEL + h * HEAD_DIM + t * 16 + col] = val;
        }
    }
}

// ---------------------------------------------------------------------------
// Host launcher
// ---------------------------------------------------------------------------
extern "C" void kernel_launch(void* const* d_in, const int* in_sizes, int n_in,
                              void* d_out, int out_size, void* d_ws, size_t ws_size,
                              hipStream_t stream) {
    const float* q   = (const float*)d_in[0];
    const float* k   = (const float*)d_in[1];
    const int*   mask = (const int*)d_in[2];
    const float* pos = (const float*)d_in[3];
    const float* Wq  = (const float*)d_in[4];
    const float* Wkv = (const float*)d_in[5];
    const float* Wr  = (const float*)d_in[6];
    const float* rrb = (const float*)d_in[7];
    const float* rwb = (const float*)d_in[8];
    float* out = (float*)d_out;

    // workspace carve (bf16 buffers), 256B aligned
    size_t off = 0;
    auto carve = [&](size_t bytes) -> char* {
        char* p = (char*)d_ws + off;
        off += (bytes + 255) & ~(size_t)255;
        return p;
    };
    unsigned short* qbf    = (unsigned short*)carve((size_t)M_TOTAL * D_MODEL * 2);
    unsigned short* kbf    = (unsigned short*)carve((size_t)M_TOTAL * D_MODEL * 2);
    unsigned short* WqT    = (unsigned short*)carve((size_t)D_MODEL * D_MODEL * 2);
    unsigned short* WkvT   = (unsigned short*)carve((size_t)D_MODEL * 2 * D_MODEL * 2);
    unsigned short* rbf    = (unsigned short*)carve((size_t)R_ROWS * HEAD_DIM * 2);
    unsigned short* qac    = (unsigned short*)carve((size_t)M_TOTAL * D_MODEL * 2);
    unsigned short* qbd    = (unsigned short*)carve((size_t)M_TOTAL * D_MODEL * 2);
    unsigned short* khb    = (unsigned short*)carve((size_t)M_TOTAL * D_MODEL * 2);
    unsigned short* vhbT   = (unsigned short*)carve((size_t)M_TOTAL * D_MODEL * 2);

    // 1) f32 -> bf16 conversions (weights transposed to N-major)
    {
        int n = M_TOTAL * D_MODEL;
        cvt_bf16_kernel<<<(n + 255) / 256, 256, 0, stream>>>(q, qbf, n);
        cvt_bf16_kernel<<<(n + 255) / 256, 256, 0, stream>>>(k, kbf, n);
        int nw = D_MODEL * D_MODEL;
        cvt_bf16_transpose_kernel<<<(nw + 255) / 256, 256, 0, stream>>>(
            Wq, WqT, D_MODEL, D_MODEL);
        int nk = D_MODEL * 2 * D_MODEL;
        cvt_bf16_transpose_kernel<<<(nk + 255) / 256, 256, 0, stream>>>(
            Wkv, WkvT, D_MODEL, 2 * D_MODEL);
    }
    // 2) r projection (+ zero pad rows)
    {
        int n = R_ROWS * HEAD_DIM;
        r_proj_kernel<<<(n + 255) / 256, 256, 0, stream>>>(pos, Wr, rbf);
    }
    // 3) q projection with biases
    proj_q_kernel<<<dim3(M_TOTAL / 16, D_MODEL / 64), 32, 0, stream>>>(
        qbf, WqT, rrb, rwb, qac, qbd);
    // 4) k/v projection (v transposed)
    proj_kv_kernel<<<dim3(M_TOTAL / 16, 2 * D_MODEL / 64), 32, 0, stream>>>(
        kbf, WkvT, khb, vhbT);
    // 5) flash attention with relative shift
    attn_kernel<<<dim3(MAX_LEN / 16, N_HEAD, BATCH), 32, SM_TOTAL, stream>>>(
        qac, qbd, khb, vhbT, rbf, mask, out);
}
